// MultiheadAttention_63153199120435
// MI455X (gfx1250) — compile-verified, matrix operand order checked by reading, not run on hardware
//
#include <hip/hip_runtime.h>
#include <hip/hip_bf16.h>
#include <cstdint>

// B=8, S=1024, MD=1024, H=16, D=64, SCALE = (D/H)^-0.5 = 0.5
#define MD_   1024
#define ELEMS 8388608   // 8*1024*1024 tokens*features

typedef __attribute__((ext_vector_type(16))) __bf16 v16bf;
typedef __attribute__((ext_vector_type(8)))  float  v8f;

__device__ __forceinline__ unsigned short f2bf(float f) {
  union { float f; unsigned u; } v; v.f = f;
  unsigned r = v.u + 0x7FFFu + ((v.u >> 16) & 1u);   // round-to-nearest-even
  return (unsigned short)(r >> 16);
}

union FragBF { v16bf v; int4 i[2]; };

// Load a 16x32 bf16 WMMA fragment (A layout; also B layout when the source
// matrix is stored transposed). base -> element [row0][k0], stride in elems.
// Lane l<16: row=l, k-chunks {0..7,16..23}; lane>=16: row=l-16, {8..15,24..31}.
__device__ __forceinline__ v16bf load_frag(const unsigned short* base, int stride, int lane) {
  int r  = lane & 15;
  int ko = (lane & 16) ? 8 : 0;
  const unsigned short* p = base + (size_t)r * stride + ko;
  FragBF f;
  f.i[0] = *reinterpret_cast<const int4*>(p);
  f.i[1] = *reinterpret_cast<const int4*>(p + 16);
  return f.v;
}

__device__ __forceinline__ v8f wmma_bf16(v16bf a, v16bf b, v8f c) {
  return __builtin_amdgcn_wmma_f32_16x16x32_bf16(false, a, false, b, (short)0, c, false, false);
}

// ---------------------------------------------------------------- converts
__global__ void k_cvt_inputs(const float* __restrict__ q, const float* __restrict__ k,
                             const float* __restrict__ v, unsigned short* __restrict__ Xb) {
  int i = blockIdx.x * blockDim.x + threadIdx.x;
  Xb[i]             = f2bf(q[i]);
  Xb[ELEMS + i]     = f2bf(k[i]);
  Xb[2 * ELEMS + i] = f2bf(v[i]);
}

// transpose + convert the four 1024x1024 weight matrices (W^T for GEMM B frags)
__global__ void k_cvt_weights(const float* __restrict__ Wq, const float* __restrict__ Wk,
                              const float* __restrict__ Wv, const float* __restrict__ Wo,
                              unsigned short* __restrict__ Wt4) {
  int i   = blockIdx.x * blockDim.x + threadIdx.x;   // coalesced read index
  int mat = blockIdx.y;
  const float* W = (mat == 0) ? Wq : (mat == 1) ? Wk : (mat == 2) ? Wv : Wo;
  int kk = i >> 10, nn = i & 1023;
  Wt4[(size_t)mat * MD_ * MD_ + (size_t)nn * MD_ + kk] = f2bf(W[i]);
}

// ---------------------------------------------------------------- QKV GEMM
// C[8192x1024] = X @ W + b ; wave computes a 16x64 tile. V is written
// transposed per-head: Vt[bh][d][s] so attention B-fragments load contiguously.
__global__ void __launch_bounds__(128)
k_qkv_gemm(const unsigned short* __restrict__ Xb, const unsigned short* __restrict__ Wt4,
           const float* __restrict__ bq, const float* __restrict__ bk, const float* __restrict__ bv,
           unsigned short* __restrict__ Qb, unsigned short* __restrict__ Kb,
           unsigned short* __restrict__ Vt) {
  int lane = threadIdx.x & 31;
  int wave = threadIdx.x >> 5;
  int m0   = blockIdx.x * 16;
  int n0   = blockIdx.y * 256 + wave * 64;
  int mat  = blockIdx.z;
  const unsigned short* X  = Xb  + (size_t)mat * ELEMS;
  const unsigned short* Wt = Wt4 + (size_t)mat * MD_ * MD_;
  const float* bias = (mat == 0) ? bq : (mat == 1) ? bk : bv;

  v8f acc[4];
#pragma unroll
  for (int nt = 0; nt < 4; ++nt) {
    float bvv = bias[n0 + nt * 16 + (lane & 15)];
#pragma unroll
    for (int r = 0; r < 8; ++r) acc[nt][r] = bvv;
  }
  for (int k0 = 0; k0 < MD_; k0 += 32) {
    v16bf a = load_frag(X + (size_t)m0 * MD_ + k0, MD_, lane);
#pragma unroll
    for (int nt = 0; nt < 4; ++nt) {
      v16bf b = load_frag(Wt + (size_t)(n0 + nt * 16) * MD_ + k0, MD_, lane);
      acc[nt] = wmma_bf16(a, b, acc[nt]);
    }
  }
  int rbase = m0 + ((lane & 16) ? 8 : 0);
#pragma unroll
  for (int nt = 0; nt < 4; ++nt) {
    int n = n0 + nt * 16 + (lane & 15);
#pragma unroll
    for (int r = 0; r < 8; ++r) {
      int row = rbase + r;
      unsigned short hv = f2bf(acc[nt][r]);
      if (mat == 0) {
        Qb[(size_t)row * MD_ + n] = hv;
      } else if (mat == 1) {
        Kb[(size_t)row * MD_ + n] = hv;
      } else {  // store V transposed per head: flat -> (bh, t, d) -> Vt[bh][d][t]
        size_t flat = (size_t)row * MD_ + n;
        int bh = (int)(flat >> 16);
        int t  = (int)((flat >> 6) & 1023);
        int d  = (int)(flat & 63);
        Vt[(size_t)bh * 65536 + (size_t)d * 1024 + t] = hv;
      }
    }
  }
}

// ---------------------------------------------------------------- attention
// One workgroup = all 16 heads of (batch b, 16 query rows). One wave per head.
// Single-pass flash attention with online softmax; K/V read exactly once.
// Sigmoid sums reduced across heads in LDS (ds_add_f32), written once.
__global__ void __launch_bounds__(512)
k_attention(const unsigned short* __restrict__ Qb, const unsigned short* __restrict__ Kb,
            const unsigned short* __restrict__ Vt, unsigned short* __restrict__ Ctx,
            float* __restrict__ attn_post) {
  __shared__ float sacc[16 * 1024];                                  // 64 KB
  __shared__ __align__(16) unsigned short pstage[16 * 16 * 40];      // 20 KB
  int lane = threadIdx.x & 31;
  int h    = threadIdx.x >> 5;
  int b    = blockIdx.x;
  int t0   = blockIdx.y * 16;
  int bh   = b * 16 + h;

  for (int i = threadIdx.x; i < 16 * 1024; i += 512) sacc[i] = 0.f;
  __syncthreads();

  const unsigned short* Qhb = Qb + (size_t)bh * 65536 + (size_t)t0 * 64;
  const unsigned short* Khb = Kb + (size_t)bh * 65536;
  const unsigned short* Vhb = Vt + (size_t)bh * 65536;

  v16bf a0 = load_frag(Qhb + 0,  64, lane);
  v16bf a1 = load_frag(Qhb + 32, 64, lane);

  v8f m, l;
  v8f o[4];
#pragma unroll
  for (int r = 0; r < 8; ++r) { m[r] = -1e30f; l[r] = 0.f; }
#pragma unroll
  for (int nt = 0; nt < 4; ++nt)
#pragma unroll
    for (int r = 0; r < 8; ++r) o[nt][r] = 0.f;

  int row_in = (lane & 16) ? 8 : 0;
  int col    = lane & 15;
  unsigned short* ps = pstage + h * (16 * 40);

  for (int s0 = 0; s0 < 1024; s0 += 32) {
    // ---- scores for two 16-wide tiles (computed once) ----
    v16bf k00 = load_frag(Khb + (size_t)s0 * 64 + 0,    64, lane);
    v16bf k01 = load_frag(Khb + (size_t)s0 * 64 + 32,   64, lane);
    v16bf k10 = load_frag(Khb + (size_t)(s0 + 16) * 64 + 0,  64, lane);
    v16bf k11 = load_frag(Khb + (size_t)(s0 + 16) * 64 + 32, 64, lane);
    v8f z = {0.f, 0.f, 0.f, 0.f, 0.f, 0.f, 0.f, 0.f};
    v8f c0 = wmma_bf16(a1, k01, wmma_bf16(a0, k00, z));
    v8f c1 = wmma_bf16(a1, k11, wmma_bf16(a0, k10, z));

    // ---- online softmax stats + sigmoid accumulation + P staging ----
    v8f sc;   // per-row ctx rescale factor exp(m_old - m_new)
#pragma unroll
    for (int r = 0; r < 8; ++r) {
      float s0v = c0[r] * 0.5f;                              // SCALE
      float s1v = c1[r] * 0.5f;
      float g0 = 1.f / (1.f + __expf(-s0v));
      float g1 = 1.f / (1.f + __expf(-s1v));
      atomicAdd(&sacc[(row_in + r) * 1024 + s0 + col],      g0);   // ds_add_f32
      atomicAdd(&sacc[(row_in + r) * 1024 + s0 + 16 + col], g1);
      // combined row-max over both tiles and the 16 lanes of this half
      float x = fmaxf(s0v, s1v);
      x = fmaxf(x, __shfl_xor(x, 1, 32));
      x = fmaxf(x, __shfl_xor(x, 2, 32));
      x = fmaxf(x, __shfl_xor(x, 4, 32));
      x = fmaxf(x, __shfl_xor(x, 8, 32));
      float mn = fmaxf(m[r], x);
      sc[r]    = __expf(m[r] - mn);
      float p0 = __expf(s0v - mn);                           // unnormalized
      float p1 = __expf(s1v - mn);
      float e  = p0 + p1;
      e += __shfl_xor(e, 1, 32);
      e += __shfl_xor(e, 2, 32);
      e += __shfl_xor(e, 4, 32);
      e += __shfl_xor(e, 8, 32);
      l[r] = l[r] * sc[r] + e;
      m[r] = mn;
      ps[(row_in + r) * 40 + col]      = f2bf(p0);
      ps[(row_in + r) * 40 + 16 + col] = f2bf(p1);
    }
    // ---- rescale running ctx, then accumulate P @ V ----
#pragma unroll
    for (int nt = 0; nt < 4; ++nt)
#pragma unroll
      for (int r = 0; r < 8; ++r) o[nt][r] *= sc[r];

    v16bf pa = load_frag(ps, 40, lane);  // 16x32 P tile in A layout
#pragma unroll
    for (int nt = 0; nt < 4; ++nt) {
      v16bf bv = load_frag(Vhb + (size_t)(nt * 16) * 1024 + s0, 1024, lane);
      o[nt] = wmma_bf16(pa, bv, o[nt]);
    }
  }

  // ---- epilogue: normalize by 1/l, store ctx (bf16) at flat [bh][t][d] ----
  v8f rl;
#pragma unroll
  for (int r = 0; r < 8; ++r) rl[r] = 1.f / l[r];
#pragma unroll
  for (int nt = 0; nt < 4; ++nt)
#pragma unroll
    for (int r = 0; r < 8; ++r) {
      int t = t0 + row_in + r;
      int d = nt * 16 + col;
      Ctx[(size_t)bh * 65536 + (size_t)t * 64 + d] = f2bf(o[nt][r] * rl[r]);
    }
  __syncthreads();
  float* outp = attn_post + (size_t)b * (1024 * 1024) + (size_t)t0 * 1024;
  for (int i = threadIdx.x; i < 16 * 1024; i += 512)
    outp[i] = sacc[i] * (1.0f / 16.0f);
}

// ---------------------------------------------------------------- out GEMM
__global__ void __launch_bounds__(128)
k_out_gemm(const unsigned short* __restrict__ Ctx, const unsigned short* __restrict__ Wot,
           const float* __restrict__ bo, float* __restrict__ Out) {
  int lane = threadIdx.x & 31;
  int wave = threadIdx.x >> 5;
  int m0   = blockIdx.x * 16;
  int n0   = blockIdx.y * 256 + wave * 64;

  v8f acc[4];
#pragma unroll
  for (int nt = 0; nt < 4; ++nt) {
    float bvv = bo[n0 + nt * 16 + (lane & 15)];
#pragma unroll
    for (int r = 0; r < 8; ++r) acc[nt][r] = bvv;
  }
  for (int k0 = 0; k0 < MD_; k0 += 32) {
    v16bf a = load_frag(Ctx + (size_t)m0 * MD_ + k0, MD_, lane);
#pragma unroll
    for (int nt = 0; nt < 4; ++nt) {
      v16bf b = load_frag(Wot + (size_t)(n0 + nt * 16) * MD_ + k0, MD_, lane);
      acc[nt] = wmma_bf16(a, b, acc[nt]);
    }
  }
  int rbase = m0 + ((lane & 16) ? 8 : 0);
#pragma unroll
  for (int nt = 0; nt < 4; ++nt) {
    int n = n0 + nt * 16 + (lane & 15);
#pragma unroll
    for (int r = 0; r < 8; ++r)
      Out[(size_t)(rbase + r) * MD_ + n] = acc[nt][r];
  }
}

extern "C" void kernel_launch(void* const* d_in, const int* in_sizes, int n_in,
                              void* d_out, int out_size, void* d_ws, size_t ws_size,
                              hipStream_t stream) {
  const float* q  = (const float*)d_in[0];
  const float* k  = (const float*)d_in[1];
  const float* v  = (const float*)d_in[2];
  const float* Wq = (const float*)d_in[3];
  const float* bq = (const float*)d_in[4];
  const float* Wk = (const float*)d_in[5];
  const float* bk = (const float*)d_in[6];
  const float* Wv = (const float*)d_in[7];
  const float* bv = (const float*)d_in[8];
  const float* Wo = (const float*)d_in[9];
  const float* bo = (const float*)d_in[10];
  float* out = (float*)d_out;

  unsigned short* ws  = (unsigned short*)d_ws;
  unsigned short* Xb  = ws;                                  // 3*ELEMS bf16
  unsigned short* Wt4 = Xb + (size_t)3 * ELEMS;              // 4*1M bf16 (W^T)
  unsigned short* Qb  = Wt4 + (size_t)4 * MD_ * MD_;         // ELEMS
  unsigned short* Kb  = Qb + (size_t)ELEMS;                  // ELEMS
  unsigned short* Vt  = Kb + (size_t)ELEMS;                  // ELEMS (V^T per head)
  unsigned short* Ctx = Vt + (size_t)ELEMS;                  // ELEMS

  k_cvt_inputs<<<ELEMS / 256, 256, 0, stream>>>(q, k, v, Xb);
  k_cvt_weights<<<dim3(4096, 4), 256, 0, stream>>>(Wq, Wk, Wv, Wo, Wt4);
  k_qkv_gemm<<<dim3(512, 4, 3), 128, 0, stream>>>(Xb, Wt4, bq, bk, bv, Qb, Kb, Vt);
  k_attention<<<dim3(8, 64), 512, 0, stream>>>(Qb, Kb, Vt, Ctx, out + (size_t)ELEMS);
  k_out_gemm<<<dim3(512, 4), 128, 0, stream>>>(Ctx, Wt4 + (size_t)3 * MD_ * MD_, bo, out);
}
